// Block_12283606468097
// MI455X (gfx1250) — matrix operand order repaired and checked by reading, not compiled, not run on hardware
//
#include <hip/hip_runtime.h>
#include <hip/hip_bf16.h>
#include <math.h>

// ---------------------------------------------------------------------------
// Transformer block (B=128, T=256, C=256, H=8, HS=32) for gfx1250 (CDNA5).
// bf16 WMMA matmuls (v_wmma_f32_16x16x32_bf16), TDM (tensor_load_to_lds)
// double-buffered tile staging with LDS padding, fp32 VALU for LN/softmax.
// ---------------------------------------------------------------------------

#define BB 128
#define TT 256
#define CC 256
#define HH 8
#define HS 32
#define MTOK (BB * TT)        // 32768 rows
#define FF (4 * CC)           // 1024

typedef __attribute__((ext_vector_type(16))) __bf16 v16bf;
typedef __attribute__((ext_vector_type(8)))  __bf16 v8bf;
typedef __attribute__((ext_vector_type(8)))  float  v8f;
typedef __attribute__((ext_vector_type(4)))  unsigned int u32x4;
typedef __attribute__((ext_vector_type(8)))  int i32x8;
typedef __attribute__((ext_vector_type(4)))  int i32x4;

// LDS row stride (elements) after TDM padding: 32 bf16 (64B) + 16B pad = 40
#define LROW 40

// Load a 16-element bf16 A/B fragment as two 16-byte chunks.
__device__ __forceinline__ v16bf ld_frag16(const __bf16* p0, const __bf16* p1) {
  v8bf lo = *(const v8bf*)p0;
  v8bf hi = *(const v8bf*)p1;
  v16bf r;
#pragma unroll
  for (int i = 0; i < 8; ++i) { r[i] = lo[i]; r[8 + i] = hi[i]; }
  return r;
}

__device__ __forceinline__ v8f zero8() {
  v8f z;
#pragma unroll
  for (int i = 0; i < 8; ++i) z[i] = 0.0f;
  return z;
}

// ---------------------------------------------------------------------------
// TDM: issue a 2D bf16 tile load (tile_k x tile_rows, row stride in elements)
// from global to LDS, with LDS padding of 16B per 64B row.
// D# layout per CDNA5 ISA ch.8 (group0 128b, group1 256b; groups 2/3 zero).
// 6-arg builtin variant (clang-23 / therock headers).
// ---------------------------------------------------------------------------
__device__ __forceinline__ void tdm_load_2d(unsigned lds_addr, const void* gptr,
                                            unsigned tile_k, unsigned tile_rows,
                                            unsigned row_stride_elems) {
  unsigned long long ga = (unsigned long long)(size_t)gptr;
  u32x4 g0;
  g0[0] = 1u;                                           // count=1 (valid D#)
  g0[1] = lds_addr;                                     // LDS byte address
  g0[2] = (unsigned)(ga & 0xFFFFFFFFu);                 // global_addr lo
  g0[3] = (unsigned)((ga >> 32) & 0x1FFFFFFu)           // global_addr hi (56:32)
        | (2u << 30);                                   // type=2 ("image")
  i32x8 g1;
  unsigned w0 = (1u << 16)                              // data_size=1 -> 2 bytes
              | (1u << 20)                              // pad_enable
              | (3u << 22)                              // pad_interval: 16 DWORDs (64B)
              | (3u << 25);                             // pad_amount: 4 DWORDs (16B)
  g1[0] = (int)w0;
  unsigned td0 = row_stride_elems;                      // tensor_dim0 (>= tile_k)
  unsigned td1 = 0x40000000u;                           // tensor_dim1: huge (no clip)
  g1[1] = (int)((td0 & 0xFFFFu) << 16);                 // [63:48] tensor_dim0 lo16
  g1[2] = (int)(((td0 >> 16) & 0xFFFFu) | ((td1 & 0xFFFFu) << 16));
  g1[3] = (int)(((td1 >> 16) & 0xFFFFu) | ((tile_k & 0xFFFFu) << 16)); // tile_dim0
  g1[4] = (int)(tile_rows & 0xFFFFu);                   // tile_dim1 (tile_dim2=0)
  g1[5] = (int)row_stride_elems;                        // tensor_dim0_stride lo32
  g1[6] = 0;                                            // stride hi16 / dim1_stride
  g1[7] = 0;
  i32x4 z4;
  z4[0] = 0; z4[1] = 0; z4[2] = 0; z4[3] = 0;
  i32x8 z8;
#pragma unroll
  for (int i = 0; i < 8; ++i) z8[i] = 0;
  __builtin_amdgcn_tensor_load_to_lds(g0, g1, z4, z4, z8, 0);
}

// ---------------------------------------------------------------------------
// Weight repack (transposed for TDM-friendly [N][K] B operands):
// Wk/Wq/Wv (H,C,HS) -> bf16 WcatT [3C][C]: row = seg*C + h*HS + d, col = c
// ---------------------------------------------------------------------------
__global__ void pack_qkv_kernel(const float* __restrict__ Wk,
                                const float* __restrict__ Wq,
                                const float* __restrict__ Wv,
                                __bf16* __restrict__ WcatT) {
  int idx = blockIdx.x * blockDim.x + threadIdx.x;   // H*C*HS = 65536
  if (idx >= HH * CC * HS) return;
  int d = idx % HS;
  int c = (idx / HS) % CC;
  int h = idx / (HS * CC);
  int col = h * HS + d;
  WcatT[(size_t)(0 * CC + col) * CC + c] = (__bf16)Wk[idx];
  WcatT[(size_t)(1 * CC + col) * CC + c] = (__bf16)Wq[idx];
  WcatT[(size_t)(2 * CC + col) * CC + c] = (__bf16)Wv[idx];
}

// src [R][Ncol] fp32 -> dst [Ncol][R] bf16 (transpose + convert)
__global__ void cvtT_bf16_kernel(const float* __restrict__ src,
                                 __bf16* __restrict__ dst, int R, int Ncol) {
  int idx = blockIdx.x * blockDim.x + threadIdx.x;
  if (idx >= R * Ncol) return;
  int r = idx / Ncol, n = idx % Ncol;
  dst[(size_t)n * R + r] = (__bf16)src[idx];
}

// ---------------------------------------------------------------------------
// LayerNorm fp32 -> bf16. One wave per row of 256; 8 rows per block.
// ---------------------------------------------------------------------------
__global__ void ln_bf16_kernel(const float* __restrict__ x,
                               const float* __restrict__ gamma,
                               const float* __restrict__ beta,
                               __bf16* __restrict__ out) {
  int wave = threadIdx.x >> 5, lane = threadIdx.x & 31;
  int row = blockIdx.x * 8 + wave;
  const float* xr = x + (size_t)row * CC;
  float v[8];
  float s = 0.0f;
#pragma unroll
  for (int i = 0; i < 8; ++i) { v[i] = xr[lane + 32 * i]; s += v[i]; }
#pragma unroll
  for (int m = 1; m < 32; m <<= 1) s += __shfl_xor(s, m, 32);
  float mu = s * (1.0f / CC);
  float vs = 0.0f;
#pragma unroll
  for (int i = 0; i < 8; ++i) { float d = v[i] - mu; vs += d * d; }
#pragma unroll
  for (int m = 1; m < 32; m <<= 1) vs += __shfl_xor(vs, m, 32);
  float rinv = rsqrtf(vs * (1.0f / CC) + 1e-5f);
  __bf16* o = out + (size_t)row * CC;
#pragma unroll
  for (int i = 0; i < 8; ++i) {
    int c = lane + 32 * i;
    o[c] = (__bf16)(gamma[c] * (v[i] - mu) * rinv + beta[c]);
  }
}

// ---------------------------------------------------------------------------
// bf16 WMMA GEMM with TDM double-buffered staging.
//   out[M,N] = A[M,K] @ Wt[N,K]^T (+bias) (opt swish) (+resid)
// Block tile 128x128, K-step 32. 8 waves as 2(M) x 4(N); wave tile 64x32
// -> 4x2 fragments = 8 x v_wmma_f32_16x16x32_bf16 per K-step.
// Wave 0 issues tensor_load_to_lds for both tiles of the next K-step, then
// s_wait_tensorcnt; TDM pads LDS rows to 80B (bank-conflict-free frags).
// ---------------------------------------------------------------------------
template <bool OUT_BF16, bool SWISH, bool RESID>
__global__ __launch_bounds__(256)
void gemm_bf16_kernel(const __bf16* __restrict__ A,
                      const __bf16* __restrict__ Wt,
                      void* __restrict__ outp,
                      const float* __restrict__ bias,
                      const float* __restrict__ resid,
                      int M, int N, int K) {
  __shared__ __bf16 As[2][128][LROW];   // 2 x 10 KB
  __shared__ __bf16 Bs[2][128][LROW];   // 2 x 10 KB

  const int tid  = threadIdx.x;
  const int wave = tid >> 5;
  const int lane = tid & 31;
  const int wm = wave >> 2;             // 0..1
  const int wn = wave & 3;              // 0..3
  const int mo = wm * 64, no = wn * 32;
  const int mblk = blockIdx.y * 128;
  const int nblk = blockIdx.x * 128;
  const int lhi = lane >> 4;            // 0/1
  const int llo = lane & 15;

  v8f acc[4][2];
#pragma unroll
  for (int a = 0; a < 4; ++a)
#pragma unroll
    for (int b = 0; b < 2; ++b) acc[a][b] = zero8();

  const int ksteps = K >> 5;
  if (wave == 0) {
    tdm_load_2d((unsigned)(size_t)&As[0][0][0],
                A + (size_t)mblk * K, 32, 128, (unsigned)K);
    tdm_load_2d((unsigned)(size_t)&Bs[0][0][0],
                Wt + (size_t)nblk * K, 32, 128, (unsigned)K);
  }

  for (int kt = 0; kt < ksteps; ++kt) {
    const int cur = kt & 1;
    if (wave == 0) {
      if (kt + 1 < ksteps) {
        tdm_load_2d((unsigned)(size_t)&As[cur ^ 1][0][0],
                    A + (size_t)mblk * K + (kt + 1) * 32, 32, 128, (unsigned)K);
        tdm_load_2d((unsigned)(size_t)&Bs[cur ^ 1][0][0],
                    Wt + (size_t)nblk * K + (kt + 1) * 32, 32, 128, (unsigned)K);
        __builtin_amdgcn_s_wait_tensorcnt(2);   // current tile pair complete
      } else {
        __builtin_amdgcn_s_wait_tensorcnt(0);
      }
    }
    __syncthreads();                            // tile kt visible to all waves

    v16bf aF[4], bF[2];
#pragma unroll
    for (int mt = 0; mt < 4; ++mt) {
      int m = mo + mt * 16 + llo;
      aF[mt] = ld_frag16(&As[cur][m][lhi * 8], &As[cur][m][16 + lhi * 8]);
    }
#pragma unroll
    for (int nt = 0; nt < 2; ++nt) {
      int n = no + nt * 16 + llo;
      bF[nt] = ld_frag16(&Bs[cur][n][lhi * 8], &Bs[cur][n][16 + lhi * 8]);
    }
#pragma unroll
    for (int mt = 0; mt < 4; ++mt)
#pragma unroll
      for (int nt = 0; nt < 2; ++nt)
        acc[mt][nt] = __builtin_amdgcn_wmma_f32_16x16x32_bf16(
            false, aF[mt], false, bF[nt], (short)0, acc[mt][nt], false, false);
    __syncthreads();                            // done reading before overwrite
  }

  // epilogue: C/D layout: lane -> (m = lhi*8 + r, n = llo)
#pragma unroll
  for (int mt = 0; mt < 4; ++mt) {
#pragma unroll
    for (int nt = 0; nt < 2; ++nt) {
      int n = nblk + no + nt * 16 + llo;
      float bv = bias ? bias[n] : 0.0f;
#pragma unroll
      for (int r = 0; r < 8; ++r) {
        int m = mblk + mo + mt * 16 + lhi * 8 + r;
        float v = acc[mt][nt][r] + bv;
        if (SWISH) v = v * (1.0f / (1.0f + __expf(-v)));
        if (RESID) v += resid[(size_t)m * N + n];
        if (OUT_BF16)
          ((__bf16*)outp)[(size_t)m * N + n] = (__bf16)v;
        else
          ((float*)outp)[(size_t)m * N + n] = v;
      }
    }
  }
}

// ---------------------------------------------------------------------------
// Flash-style causal attention per (b,h). scores = K·Qᵀ/√C, softmax, ·V.
// KQV layout: [B*T][3C], segments K|Q|V, cols h*HS+d. Output O bf16 [B*T][C].
// One block (8 waves) per (b,h); wave handles row tiles {w, w+8} of 16 rows.
// ---------------------------------------------------------------------------
__global__ __launch_bounds__(256)
void attn_kernel(const __bf16* __restrict__ KQV, __bf16* __restrict__ O) {
  __shared__ __bf16 Ks[TT][HS];        // 16 KB
  __shared__ __bf16 Qs[TT][HS];        // 16 KB
  __shared__ __bf16 VsT[HS][TT];       // 16 KB (transposed)
  __shared__ __bf16 Pbuf[8][16][32];   // per-wave P staging, 8 KB

  const int bh = blockIdx.x;
  const int b = bh / HH, h = bh % HH;
  const int tid = threadIdx.x, wave = tid >> 5, lane = tid & 31;
  const int lhi = lane >> 4, llo = lane & 15;
  const float scale = 0.0625f;         // C^-0.5 = 1/16
  const float NEG = -1.0e30f;

  // stage K/Q/V rows: one row per thread
  {
    const __bf16* src = KQV + (size_t)(b * TT + tid) * (3 * CC) + h * HS;
#pragma unroll
    for (int c = 0; c < 4; ++c) {
      v8bf kk = *(const v8bf*)(src + 0 * CC + c * 8);
      v8bf qq = *(const v8bf*)(src + 1 * CC + c * 8);
      *(v8bf*)&Ks[tid][c * 8] = kk;
      *(v8bf*)&Qs[tid][c * 8] = qq;
      v8bf vv = *(const v8bf*)(src + 2 * CC + c * 8);
#pragma unroll
      for (int i = 0; i < 8; ++i) VsT[c * 8 + i][tid] = vv[i];
    }
  }
  __syncthreads();

  for (int rr = 0; rr < 2; ++rr) {
    const int rt = wave + rr * 8;      // row tile index 0..15
    const int mbase = rt * 16;
    const int am = mbase + llo;
    const v16bf aK = ld_frag16(&Ks[am][lhi * 8], &Ks[am][16 + lhi * 8]);

    v8f acc0 = zero8(), acc1 = zero8();
    float rmax[8], rsum[8];
#pragma unroll
    for (int r = 0; r < 8; ++r) { rmax[r] = NEG; rsum[r] = 0.0f; }

    const int cmax = (mbase + 15) >> 5;
    for (int sc = 0; sc <= cmax; ++sc) {
      const int sbase = sc * 32;
      int s0 = sbase + llo, s1 = s0 + 16;
      v16bf bQ0 = ld_frag16(&Qs[s0][lhi * 8], &Qs[s0][16 + lhi * 8]);
      v16bf bQ1 = ld_frag16(&Qs[s1][lhi * 8], &Qs[s1][16 + lhi * 8]);
      v8f Sc0 = __builtin_amdgcn_wmma_f32_16x16x32_bf16(
          false, aK, false, bQ0, (short)0, zero8(), false, false);
      v8f Sc1 = __builtin_amdgcn_wmma_f32_16x16x32_bf16(
          false, aK, false, bQ1, (short)0, zero8(), false, false);

      const int mrow0 = mbase + lhi * 8;
      const int n0 = sbase + llo, n1 = n0 + 16;
#pragma unroll
      for (int r = 0; r < 8; ++r) {
        int m = mrow0 + r;
        float v0 = (n0 <= m) ? Sc0[r] * scale : NEG;
        float v1 = (n1 <= m) ? Sc1[r] * scale : NEG;
        float tm = fmaxf(v0, v1);
#pragma unroll
        for (int mk = 1; mk < 16; mk <<= 1) tm = fmaxf(tm, __shfl_xor(tm, mk, 32));
        float nm = fmaxf(rmax[r], tm);
        float p0 = __expf(v0 - nm);
        float p1 = __expf(v1 - nm);
        float ps = p0 + p1;
#pragma unroll
        for (int mk = 1; mk < 16; mk <<= 1) ps += __shfl_xor(ps, mk, 32);
        float corr = __expf(rmax[r] - nm);
        rsum[r] = rsum[r] * corr + ps;
        acc0[r] *= corr;
        acc1[r] *= corr;
        rmax[r] = nm;
        int mloc = lhi * 8 + r;
        Pbuf[wave][mloc][llo]      = (__bf16)p0;
        Pbuf[wave][mloc][llo + 16] = (__bf16)p1;
      }
      v16bf aP = ld_frag16(&Pbuf[wave][llo][lhi * 8],
                           &Pbuf[wave][llo][16 + lhi * 8]);
      v16bf bV0 = ld_frag16(&VsT[llo][sbase + lhi * 8],
                            &VsT[llo][sbase + 16 + lhi * 8]);
      v16bf bV1 = ld_frag16(&VsT[16 + llo][sbase + lhi * 8],
                            &VsT[16 + llo][sbase + 16 + lhi * 8]);
      acc0 = __builtin_amdgcn_wmma_f32_16x16x32_bf16(
          false, aP, false, bV0, (short)0, acc0, false, false);
      acc1 = __builtin_amdgcn_wmma_f32_16x16x32_bf16(
          false, aP, false, bV1, (short)0, acc1, false, false);
    }

#pragma unroll
    for (int r = 0; r < 8; ++r) {
      float inv = 1.0f / rsum[r];
      int t = mbase + lhi * 8 + r;
      __bf16* o = O + (size_t)(b * TT + t) * CC + h * HS;
      o[llo]      = (__bf16)(acc0[r] * inv);
      o[llo + 16] = (__bf16)(acc1[r] * inv);
    }
  }
}

// ---------------------------------------------------------------------------
// Host-side launcher
// ---------------------------------------------------------------------------
extern "C" void kernel_launch(void* const* d_in, const int* in_sizes, int n_in,
                              void* d_out, int out_size, void* d_ws, size_t ws_size,
                              hipStream_t stream) {
  (void)in_sizes; (void)n_in; (void)out_size; (void)ws_size;

  const float* x   = (const float*)d_in[0];
  const float* Wk  = (const float*)d_in[1];
  const float* Wq  = (const float*)d_in[2];
  const float* Wv  = (const float*)d_in[3];
  const float* Wp  = (const float*)d_in[4];
  const float* bp  = (const float*)d_in[5];
  const float* W1  = (const float*)d_in[6];
  const float* b1  = (const float*)d_in[7];
  const float* W2  = (const float*)d_in[8];
  const float* b2  = (const float*)d_in[9];
  const float* g1  = (const float*)d_in[10];
  const float* be1 = (const float*)d_in[11];
  const float* g2  = (const float*)d_in[12];
  const float* be2 = (const float*)d_in[13];
  float* out = (float*)d_out;

  char* ws = (char*)d_ws;
  size_t off = 0;
  auto carve = [&](size_t bytes) -> void* {
    void* p = ws + off;
    off += (bytes + 255) & ~(size_t)255;
    return p;
  };
  __bf16* WcatT = (__bf16*)carve((size_t)3 * CC * CC * 2);  // [768][256]
  __bf16* WpT   = (__bf16*)carve((size_t)CC * CC * 2);      // [256][256]
  __bf16* W1T   = (__bf16*)carve((size_t)FF * CC * 2);      // [1024][256]
  __bf16* W2T   = (__bf16*)carve((size_t)CC * FF * 2);      // [256][1024]
  __bf16* h1    = (__bf16*)carve((size_t)MTOK * CC * 2);
  __bf16* KQV   = (__bf16*)carve((size_t)MTOK * 3 * CC * 2);
  __bf16* Obuf  = (__bf16*)carve((size_t)MTOK * CC * 2);
  float*  xo    = (float*) carve((size_t)MTOK * CC * 4);
  __bf16* h2    = (__bf16*)carve((size_t)MTOK * CC * 2);
  __bf16* f1s   = (__bf16*)carve((size_t)MTOK * FF * 2);

  // 1) weight conversion / packing (bf16, transposed [N][K])
  pack_qkv_kernel<<<(HH * CC * HS) / 256, 256, 0, stream>>>(Wk, Wq, Wv, WcatT);
  cvtT_bf16_kernel<<<(CC * CC) / 256, 256, 0, stream>>>(Wp, WpT, CC, CC);
  cvtT_bf16_kernel<<<(CC * FF) / 256, 256, 0, stream>>>(W1, W1T, CC, FF);
  cvtT_bf16_kernel<<<(FF * CC) / 256, 256, 0, stream>>>(W2, W2T, FF, CC);

  // 2) ln1(x) -> h1 (bf16)
  ln_bf16_kernel<<<MTOK / 8, 256, 0, stream>>>(x, g1, be1, h1);

  // 3) KQV = h1 @ WcatT^T  (M=32768, N=768, K=256), bf16 out
  gemm_bf16_kernel<true, false, false>
      <<<dim3((3 * CC) / 128, MTOK / 128), 256, 0, stream>>>(
          h1, WcatT, (void*)KQV, nullptr, nullptr, MTOK, 3 * CC, CC);

  // 4) causal attention per (b,h) -> Obuf (bf16, head-concat)
  attn_kernel<<<BB * HH, 256, 0, stream>>>(KQV, Obuf);

  // 5) xo = x + Obuf @ Wp + bp  (fp32)
  gemm_bf16_kernel<false, false, true>
      <<<dim3(CC / 128, MTOK / 128), 256, 0, stream>>>(
          Obuf, WpT, (void*)xo, bp, x, MTOK, CC, CC);

  // 6) ln2(xo) -> h2 (bf16)
  ln_bf16_kernel<<<MTOK / 8, 256, 0, stream>>>(xo, g2, be2, h2);

  // 7) f1s = swish(h2 @ W1 + b1)  (M=32768, N=1024, K=256), bf16 out
  gemm_bf16_kernel<true, true, false>
      <<<dim3(FF / 128, MTOK / 128), 256, 0, stream>>>(
          h2, W1T, (void*)f1s, b1, nullptr, MTOK, FF, CC);

  // 8) out = xo + f1s @ W2 + b2  (M=32768, N=256, K=1024), fp32 out
  gemm_bf16_kernel<false, false, true>
      <<<dim3(CC / 128, MTOK / 128), 256, 0, stream>>>(
          f1s, W2T, (void*)out, b2, xo, MTOK, CC, FF);
}
